// Attention_3126736192219
// MI455X (gfx1250) — compile-verified, matrix operand order checked by reading, not run on hardware
//
#include <hip/hip_runtime.h>
#include <hip/hip_bf16.h>

typedef __bf16 bf16;
typedef __attribute__((ext_vector_type(16))) __bf16 v16bf;
typedef __attribute__((ext_vector_type(8)))  __bf16 v8bf;
typedef __attribute__((ext_vector_type(8)))  float  v8f;

#define S_CTX 2048
#define NHEAD 16
#define DMODEL 1024
#define DHEAD 64

__device__ __forceinline__ v16bf combine16(v8bf lo, v8bf hi) {
  return __builtin_shufflevector(lo, hi, 0,1,2,3,4,5,6,7,8,9,10,11,12,13,14,15);
}

__device__ __forceinline__ v8f bf_wmma(v16bf a, v16bf b, v8f c) {
  return __builtin_amdgcn_wmma_f32_16x16x32_bf16(false, a, false, b, (short)0, c, false, false);
}

// Load two contiguous 8-float runs and pack to a 16-element bf16 A fragment.
__device__ __forceinline__ v16bf pack_a(const float* p1, const float* p2) {
  float4 f0 = *(const float4*)(p1);
  float4 f1 = *(const float4*)(p1 + 4);
  float4 f2 = *(const float4*)(p2);
  float4 f3 = *(const float4*)(p2 + 4);
  v16bf a;
  a[0]=(bf16)f0.x;  a[1]=(bf16)f0.y;  a[2]=(bf16)f0.z;  a[3]=(bf16)f0.w;
  a[4]=(bf16)f1.x;  a[5]=(bf16)f1.y;  a[6]=(bf16)f1.z;  a[7]=(bf16)f1.w;
  a[8]=(bf16)f2.x;  a[9]=(bf16)f2.y;  a[10]=(bf16)f2.z; a[11]=(bf16)f2.w;
  a[12]=(bf16)f3.x; a[13]=(bf16)f3.y; a[14]=(bf16)f3.z; a[15]=(bf16)f3.w;
  return a;
}

// Map a K-offset within a 32-chunk to (khalf, element index j) per ISA 16-bit A/B layout.
__device__ __forceinline__ void koff_map(int koff, int& h, int& j) {
  if (koff < 8)       { h = 0; j = koff;      }
  else if (koff < 16) { h = 1; j = koff - 8;  }
  else if (koff < 24) { h = 0; j = koff - 8;  }
  else                { h = 1; j = koff - 16; }
}

__device__ __forceinline__ float rowmax16(float v) {
  v = fmaxf(v, __shfl_xor(v, 1, 32));
  v = fmaxf(v, __shfl_xor(v, 2, 32));
  v = fmaxf(v, __shfl_xor(v, 4, 32));
  v = fmaxf(v, __shfl_xor(v, 8, 32));
  return v;
}
__device__ __forceinline__ float rowsum16(float v) {
  v += __shfl_xor(v, 1, 32);
  v += __shfl_xor(v, 2, 32);
  v += __shfl_xor(v, 4, 32);
  v += __shfl_xor(v, 8, 32);
  return v;
}

// ---------------------------------------------------------------------------
// Kernel 1: q/k/v projection (per head GEMM 2048x1024 @ 1024x64) + bias +
// RoPE (q,k) + 1/ATTN_SCALE folded into q. Outputs bf16.
// Block = (head, sel in {q,k,v}, column-pair p, row-group rg). Column pair =
// cols [16p,16p+16) and [16p+32,16p+48) so RoPE partners live in one block.
// 4 row-groups of 32 row tiles -> 384 blocks for WGP occupancy.
// ---------------------------------------------------------------------------
__global__ __launch_bounds__(256) void qkv_rope_kernel(
    const float* __restrict__ xq, const float* __restrict__ xk, const float* __restrict__ xv,
    const float* __restrict__ WQ, const float* __restrict__ bQ,
    const float* __restrict__ WK, const float* __restrict__ bK,
    const float* __restrict__ WV, const float* __restrict__ bV,
    bf16* __restrict__ qws, bf16* __restrict__ kws, bf16* __restrict__ vws)
{
  __shared__ bf16 Wlds[32 * 32 * 2 * 16];  // 64 KB, fragment-swizzled

  const int b   = blockIdx.x;
  const int rg  = b & 3;            // row group (0..3)
  const int c6  = b >> 2;
  const int n   = c6 / 6;
  const int rem = c6 % 6;
  const int sel = rem >> 1;         // 0=q 1=k 2=v
  const int p   = rem & 1;          // column pair

  const float* X    = (sel == 0) ? xq : (sel == 1) ? xk : xv;
  const float* W    = (sel == 0) ? WQ : (sel == 1) ? WK : WV;
  const float* bias = (sel == 0) ? bQ : (sel == 1) ? bK : bV;

  // Stage W[:, pair cols] -> LDS in per-lane-contiguous B-fragment order.
  for (int idx = threadIdx.x; idx < 32768; idx += 256) {
    int k = idx >> 5;
    int c = idx & 31;
    int e = (c < 16) ? (p * 16 + c) : (p * 16 + 32 + (c - 16));
    float w = W[((size_t)n * DMODEL + k) * DHEAD + e];
    int h, j; koff_map(k & 31, h, j);
    Wlds[((((k >> 5) * 32 + c) * 2 + h) << 4) + j] = (bf16)w;
  }
  __syncthreads();

  const int wv   = threadIdx.x >> 5;
  const int lane = threadIdx.x & 31;
  const int nloc = lane & 15;
  const int hh   = lane >> 4;
  const int e_lo = p * 16 + nloc;
  const int e_hi = e_lo + 32;
  const float inv_freq = __powf(10000.0f, -(float)e_lo * (1.0f / 32.0f));
  const float biasLo = bias[n * DHEAD + e_lo];
  const float biasHi = bias[n * DHEAD + e_hi];

  const int rt_beg = rg * 32 + wv;
  const int rt_end = rg * 32 + 32;
  for (int rt = rt_beg; rt < rt_end; rt += 8) {
    const int srow = rt * 16 + nloc;                       // A-fragment row
    const float* xrow = X + ((size_t)(srow * NHEAD + n) << 10);
    v8f accLo = {};
    v8f accHi = {};
    #pragma unroll 4
    for (int k0 = 0; k0 < DMODEL; k0 += 32) {
      v16bf a = pack_a(xrow + k0 + hh * 8, xrow + k0 + 16 + hh * 8);
      const bf16* fLo = &Wlds[((((k0 >> 5) * 32 + nloc) * 2 + hh) << 4)];
      const bf16* fHi = &Wlds[((((k0 >> 5) * 32 + 16 + nloc) * 2 + hh) << 4)];
      v16bf bLo = combine16(*(const v8bf*)fLo, *(const v8bf*)(fLo + 8));
      v16bf bHi = combine16(*(const v8bf*)fHi, *(const v8bf*)(fHi + 8));
      accLo = bf_wmma(a, bLo, accLo);
      accHi = bf_wmma(a, bHi, accHi);
    }
    #pragma unroll
    for (int r = 0; r < 8; r++) {
      int s = rt * 16 + r + hh * 8;                        // C-fragment row
      float lo = accLo[r] + biasLo;
      float hi = accHi[r] + biasHi;
      if (sel < 2) {
        float sn, cs;
        __sincosf((float)s * inv_freq, &sn, &cs);
        float rlo = lo * cs - hi * sn;
        float rhi = hi * cs + lo * sn;
        if (sel == 0) { rlo *= 0.125f; rhi *= 0.125f; }    // 1/ATTN_SCALE
        bf16* dst = (sel == 0) ? qws : kws;
        dst[((size_t)n * S_CTX + s) * DHEAD + e_lo] = (bf16)rlo;
        dst[((size_t)n * S_CTX + s) * DHEAD + e_hi] = (bf16)rhi;
      } else {
        // v stored transposed [n][e][s] so attention B-fragments are contiguous
        vws[((size_t)n * DHEAD + e_lo) * S_CTX + s] = (bf16)lo;
        vws[((size_t)n * DHEAD + e_hi) * S_CTX + s] = (bf16)hi;
      }
    }
  }
}

// ---------------------------------------------------------------------------
// Kernel 2: causal flash attention. Per 32-key step: 4 score WMMAs, online
// softmax, P re-layout via per-wave LDS, 4 z WMMAs against transposed V.
// Load balance: wave t handles q-tiles t and 127-t -> every wave does
// exactly 129 key-tiles of work.
// ---------------------------------------------------------------------------
__device__ __forceinline__ void attn_qtile(
    int n, int qt, int wv, int nloc, int hh,
    const bf16* __restrict__ qws, const bf16* __restrict__ kws,
    const bf16* __restrict__ vws, bf16* __restrict__ zws,
    bf16 (*Plds)[16 * 32])
{
  // Q A-fragments (K=64 -> two chunks), loaded once.
  const bf16* qrow = qws + ((size_t)n * S_CTX + qt * 16 + nloc) * DHEAD;
  v16bf qa0 = combine16(*(const v8bf*)(qrow + hh * 8),      *(const v8bf*)(qrow + 16 + hh * 8));
  v16bf qa1 = combine16(*(const v8bf*)(qrow + 32 + hh * 8), *(const v8bf*)(qrow + 48 + hh * 8));

  float m[8], l[8];
  v8f z0 = {}, z1 = {}, z2 = {}, z3 = {};
  #pragma unroll
  for (int r = 0; r < 8; r++) { m[r] = -3.0e38f; l[r] = 0.0f; }

  const int kv_end = (qt + 1) * 16;
  for (int kv0 = 0; kv0 < kv_end; kv0 += 32) {
    // ---- scores: two 16x16 C tiles over this 32-key strip ----
    v8f s0 = {}, s1 = {};
    {
      const bf16* kr = kws + ((size_t)n * S_CTX + kv0 + nloc) * DHEAD;
      v16bf kb0 = combine16(*(const v8bf*)(kr + hh * 8),      *(const v8bf*)(kr + 16 + hh * 8));
      v16bf kb1 = combine16(*(const v8bf*)(kr + 32 + hh * 8), *(const v8bf*)(kr + 48 + hh * 8));
      s0 = bf_wmma(qa0, kb0, s0);
      s0 = bf_wmma(qa1, kb1, s0);
    }
    {
      const bf16* kr = kws + ((size_t)n * S_CTX + kv0 + 16 + nloc) * DHEAD;
      v16bf kb0 = combine16(*(const v8bf*)(kr + hh * 8),      *(const v8bf*)(kr + 16 + hh * 8));
      v16bf kb1 = combine16(*(const v8bf*)(kr + 32 + hh * 8), *(const v8bf*)(kr + 48 + hh * 8));
      s1 = bf_wmma(qa0, kb0, s1);
      s1 = bf_wmma(qa1, kb1, s1);
    }
    // ---- causal mask + online softmax stats (row = r + 8*hh half) ----
    float p0[8], p1[8], sc[8];
    #pragma unroll
    for (int r = 0; r < 8; r++) {
      int qrow_i = qt * 16 + r + hh * 8;
      float v0 = ((kv0 + nloc)      > qrow_i) ? -3.0e38f : s0[r];
      float v1 = ((kv0 + 16 + nloc) > qrow_i) ? -3.0e38f : s1[r];
      float tmax = rowmax16(fmaxf(v0, v1));
      float mn   = fmaxf(m[r], tmax);
      float e0 = __expf(v0 - mn);
      float e1 = __expf(v1 - mn);
      float rs = rowsum16(e0 + e1);
      float scale = __expf(m[r] - mn);
      l[r] = l[r] * scale + rs;
      m[r] = mn;
      sc[r] = scale;
      p0[r] = e0; p1[r] = e1;
    }
    #pragma unroll
    for (int r = 0; r < 8; r++) {
      z0[r] *= sc[r]; z1[r] *= sc[r]; z2[r] *= sc[r]; z3[r] *= sc[r];
    }
    // ---- re-layout P: C layout -> A fragment via per-wave LDS ----
    #pragma unroll
    for (int r = 0; r < 8; r++) {
      int row = r + hh * 8;
      Plds[wv][row * 32 + nloc]      = (bf16)p0[r];
      Plds[wv][row * 32 + 16 + nloc] = (bf16)p1[r];
    }
    asm volatile("s_wait_dscnt 0" ::: "memory");
    const bf16* prow = &Plds[wv][nloc * 32];
    v16bf pa = combine16(*(const v8bf*)(prow + hh * 8), *(const v8bf*)(prow + 16 + hh * 8));
    // ---- z += P(16x32) x V(32x64), V stored [n][e][s] ----
    #pragma unroll
    for (int et = 0; et < 4; et++) {
      const bf16* vcol = vws + ((size_t)n * DHEAD + et * 16 + nloc) * S_CTX + kv0;
      v16bf vb = combine16(*(const v8bf*)(vcol + hh * 8), *(const v8bf*)(vcol + 16 + hh * 8));
      v8f& zz = (et == 0) ? z0 : (et == 1) ? z1 : (et == 2) ? z2 : z3;
      zz = bf_wmma(pa, vb, zz);
    }
  }
  // ---- normalize and store z as bf16 [n][s][e] ----
  float rl[8];
  #pragma unroll
  for (int r = 0; r < 8; r++) rl[r] = 1.0f / l[r];
  #pragma unroll
  for (int et = 0; et < 4; et++) {
    const v8f& zz = (et == 0) ? z0 : (et == 1) ? z1 : (et == 2) ? z2 : z3;
    #pragma unroll
    for (int r = 0; r < 8; r++) {
      int s = qt * 16 + r + hh * 8;
      zws[((size_t)n * S_CTX + s) * DHEAD + et * 16 + nloc] = (bf16)(zz[r] * rl[r]);
    }
  }
}

__global__ __launch_bounds__(256) void attn_kernel(
    const bf16* __restrict__ qws, const bf16* __restrict__ kws,
    const bf16* __restrict__ vws, bf16* __restrict__ zws)
{
  __shared__ bf16 Plds[8][16 * 32];  // per-wave P tile

  const int wv   = threadIdx.x >> 5;
  const int lane = threadIdx.x & 31;
  const int nloc = lane & 15;
  const int hh   = lane >> 4;
  const int task = blockIdx.x * 8 + wv;  // 0..1023
  const int n    = task >> 6;            // head
  const int t    = task & 63;

  attn_qtile(n, t,       wv, nloc, hh, qws, kws, vws, zws, Plds);
  attn_qtile(n, 127 - t, wv, nloc, hh, qws, kws, vws, zws, Plds);
}

// ---------------------------------------------------------------------------
// Kernel 3: per-head output projection z(2048x64) @ W_O[n](64x1024) + b_O/16.
// Block = (head, 64-col group, row-group). W_O fragments hoisted to registers.
// ---------------------------------------------------------------------------
__global__ __launch_bounds__(256) void outproj_kernel(
    const bf16* __restrict__ zws, const float* __restrict__ WO,
    const float* __restrict__ bO, float* __restrict__ out)
{
  __shared__ bf16 Wlds[2 * 64 * 2 * 16];  // 8 KB, fragment-swizzled

  const int rg = blockIdx.x & 1;          // row group (0..1)
  const int cb = blockIdx.x >> 1;
  const int n  = cb >> 4;
  const int d0 = (cb & 15) * 64;

  for (int idx = threadIdx.x; idx < 4096; idx += 256) {
    int k = idx >> 6;           // 0..63 (E)
    int c = idx & 63;           // 0..63 (col within group)
    float w = WO[((size_t)n * DHEAD + k) * DMODEL + d0 + c];
    int h, j; koff_map(k & 31, h, j);
    Wlds[((((k >> 5) * 64 + c) * 2 + h) << 4) + j] = (bf16)w;
  }
  __syncthreads();

  const int wv   = threadIdx.x >> 5;
  const int lane = threadIdx.x & 31;
  const int nloc = lane & 15;
  const int hh   = lane >> 4;

  v16bf wb[4][2];
  #pragma unroll
  for (int ct = 0; ct < 4; ct++)
    #pragma unroll
    for (int kc = 0; kc < 2; kc++) {
      const bf16* f = &Wlds[(((kc * 64 + ct * 16 + nloc) * 2 + hh) << 4)];
      wb[ct][kc] = combine16(*(const v8bf*)f, *(const v8bf*)(f + 8));
    }
  float bias[4];
  #pragma unroll
  for (int ct = 0; ct < 4; ct++)
    bias[ct] = bO[d0 + ct * 16 + nloc] * (1.0f / 16.0f);

  const int rt_beg = rg * 64 + wv;
  const int rt_end = rg * 64 + 64;
  for (int rt = rt_beg; rt < rt_end; rt += 8) {
    const bf16* zrow = zws + ((size_t)n * S_CTX + rt * 16 + nloc) * DHEAD;
    v16bf a0 = combine16(*(const v8bf*)(zrow + hh * 8),      *(const v8bf*)(zrow + 16 + hh * 8));
    v16bf a1 = combine16(*(const v8bf*)(zrow + 32 + hh * 8), *(const v8bf*)(zrow + 48 + hh * 8));
    #pragma unroll
    for (int ct = 0; ct < 4; ct++) {
      v8f acc = {};
      acc = bf_wmma(a0, wb[ct][0], acc);
      acc = bf_wmma(a1, wb[ct][1], acc);
      #pragma unroll
      for (int r = 0; r < 8; r++) {
        int s = rt * 16 + r + hh * 8;
        out[((size_t)(s * NHEAD + n) << 10) + d0 + ct * 16 + nloc] = acc[r] + bias[ct];
      }
    }
  }
}

// ---------------------------------------------------------------------------
extern "C" void kernel_launch(void* const* d_in, const int* in_sizes, int n_in,
                              void* d_out, int out_size, void* d_ws, size_t ws_size,
                              hipStream_t stream) {
  (void)in_sizes; (void)n_in; (void)out_size; (void)ws_size;
  const float* xq = (const float*)d_in[0];
  const float* xk = (const float*)d_in[1];
  const float* xv = (const float*)d_in[2];
  const float* WQ = (const float*)d_in[3];
  const float* bQ = (const float*)d_in[4];
  const float* WK = (const float*)d_in[5];
  const float* bK = (const float*)d_in[6];
  const float* WV = (const float*)d_in[7];
  const float* bV = (const float*)d_in[8];
  const float* WO = (const float*)d_in[9];
  const float* bO = (const float*)d_in[10];
  float* out = (float*)d_out;

  char* ws = (char*)d_ws;
  const size_t seg = (size_t)NHEAD * S_CTX * DHEAD * sizeof(bf16);  // 4 MiB
  bf16* qws = (bf16*)(ws);
  bf16* kws = (bf16*)(ws + seg);
  bf16* vws = (bf16*)(ws + 2 * seg);
  bf16* zws = (bf16*)(ws + 3 * seg);

  qkv_rope_kernel<<<dim3(NHEAD * 3 * 2 * 4), 256, 0, stream>>>(
      xq, xk, xv, WQ, bQ, WK, bK, WV, bV, qws, kws, vws);
  attn_kernel<<<dim3(128), 256, 0, stream>>>(qws, kws, vws, zws);
  outproj_kernel<<<dim3(512), 256, 0, stream>>>(zws, WO, bO, out);
}
